// Attention_27006754357835
// MI455X (gfx1250) — matrix operand order repaired and checked by reading, not compiled, not run on hardware
//
#include <hip/hip_runtime.h>
#include <hip/hip_bf16.h>
#include <math.h>

// Problem constants (reference: B,S,H = 32, 2048, 1024)
#define BB 32
#define SS 2048
#define HH 1024
#define TILE_M 32
#define LDA (HH + 8)   // bf16 elements per LDS row (pad: 8 bf16 = 4 dwords -> conflict-free)

typedef __attribute__((ext_vector_type(16))) __bf16 v16bf;
typedef __attribute__((ext_vector_type(8)))  __bf16 v8bf;
typedef __attribute__((ext_vector_type(4)))  __bf16 v4bf;
typedef __attribute__((ext_vector_type(8)))  float  v8f;

// ---------------------------------------------------------------------------
// Kernel 0: one-time W_h fp32 -> bf16 conversion into workspace.
// W_h is reused by every workgroup of the main GEMM; the bf16 copy (2 MB)
// stays L2-resident and feeds the WMMA B operand with zero per-step cvt ALU.
// ---------------------------------------------------------------------------
__global__ void wh_bf16_kernel(const float* __restrict__ W_h,
                               __bf16* __restrict__ Whb) {
    const size_t i = ((size_t)blockIdx.x * 256 + threadIdx.x) * 8;
    float4 f0 = *(const float4*)(W_h + i);
    float4 f1 = *(const float4*)(W_h + i + 4);
    v8bf q;
    q[0] = (__bf16)f0.x; q[1] = (__bf16)f0.y; q[2] = (__bf16)f0.z; q[3] = (__bf16)f0.w;
    q[4] = (__bf16)f1.x; q[5] = (__bf16)f1.y; q[6] = (__bf16)f1.z; q[7] = (__bf16)f1.w;
    *(v8bf*)(Whb + i) = q;
}

// ---------------------------------------------------------------------------
// Kernel 1: dec_feature[b,o] = dec_hidden[b,:] . W_s[o,:] + b_s[o]
// Tiny (67 MFLOP) -> simple VALU kernel. grid(B, H/256), block 256.
// ---------------------------------------------------------------------------
__global__ void dec_feature_kernel(const float* __restrict__ dec_hidden,
                                   const float* __restrict__ W_s,
                                   const float* __restrict__ b_s,
                                   float* __restrict__ dec_feat) {
    __shared__ float dh[HH];
    const int b = blockIdx.x;
    const int o = blockIdx.y * 256 + threadIdx.x;
    for (int i = threadIdx.x; i < HH; i += 256) dh[i] = dec_hidden[(size_t)b * HH + i];
    __syncthreads();
    const float* wr = W_s + (size_t)o * HH;
    float acc = b_s[o];
    for (int k = 0; k < HH; k += 4) {
        float4 w = *(const float4*)(wr + k);
        acc += dh[k] * w.x + dh[k + 1] * w.y + dh[k + 2] * w.z + dh[k + 3] * w.w;
    }
    dec_feat[(size_t)b * HH + o] = acc;
}

// ---------------------------------------------------------------------------
// Kernel 2: fused attention scores.
//   e_t[b,s] = sum_n V[n] * tanh( (enc[b,s,:] . W_h[n,:]) + dec_feat[b,n]
//                                 + coverage[b,s] * W_c[n] )
// GEMM M=B*S, N=1024, K=1024 with v_wmma_f32_16x16x32_bf16, fp32 accumulate.
// Block: 256 thr (8 wave32). Tile: 32 rows (two 16-row M-tiles), full N.
// K-loop kept ROLLED (unroll_count 4) so the LDS A-tile loads stay inside
// the loop: full unroll made LICM hoist ~512 VGPRs of A data out of the
// nt-loop and spill the whole tile to scratch (seen in round-2 asm).
// grid(S/32, B).
// ---------------------------------------------------------------------------
__global__ __launch_bounds__(256, 2)
void attn_scores_kernel(const float* __restrict__ enc,
                        const float* __restrict__ dec_feat,
                        const float* __restrict__ coverage,
                        const __bf16* __restrict__ Whb,
                        const float* __restrict__ W_c,
                        const float* __restrict__ Vw,
                        float* __restrict__ e_out) {
    __shared__ __bf16 Alds[TILE_M * LDA];
    __shared__ float  cov_lds[TILE_M];
    __shared__ float  e_part[8][TILE_M];

    const int b   = blockIdx.y;
    const int s0  = blockIdx.x * TILE_M;
    const int tid = threadIdx.x;

    // --- Stage A tile: 32x1024 contiguous fp32 -> bf16 in LDS (padded rows)
    const float* Ag = enc + ((size_t)b * SS + s0) * HH;
    for (int base = tid * 4; base < TILE_M * HH; base += 256 * 4) {
        float4 f = *(const float4*)(Ag + base);
        const int r = base / HH, c = base % HH;
        v4bf q;
        q[0] = (__bf16)f.x; q[1] = (__bf16)f.y; q[2] = (__bf16)f.z; q[3] = (__bf16)f.w;
        *(v4bf*)(&Alds[r * LDA + c]) = q;
    }
    if (tid < TILE_M) cov_lds[tid] = coverage[(size_t)b * SS + s0 + tid];
    __syncthreads();

    const int wave = tid >> 5;
    const int lane = tid & 31;
    const int mrow = lane & 15;     // N column (B/D) and M row (A) for this lane
    const int hi   = lane >> 4;     // 0: lanes 0-15, 1: lanes 16-31
    const int kcA  = hi * 16;       // A-operand K base (contiguous, matches B)
    const int kcB  = hi * 16;       // B-operand K base (contiguous, matches A)
    const int mb   = hi * 8;        // D-operand M base per ISA layout

    float cov0[8], cov1[8];
    for (int v = 0; v < 8; ++v) {
        cov0[v] = cov_lds[mb + v];
        cov1[v] = cov_lds[16 + mb + v];
    }
    float part0[8], part1[8];
    for (int v = 0; v < 8; ++v) { part0[v] = 0.f; part1[v] = 0.f; }

    const __bf16* a0p = &Alds[mrow * LDA];
    const __bf16* a1p = &Alds[(16 + mrow) * LDA];

    // Each wave handles 8 of the 64 N-tiles.
#pragma clang loop unroll(disable)
    for (int nt = wave; nt < HH / 16; nt += 8) {
        const int n = nt * 16 + mrow;                // this lane's output column
        const __bf16* wr = Whb + (size_t)n * HH;     // B[k,n] = W_h[n,k] (bf16)
        // Prefetch next n-tile's row of W_h while this K-loop runs.
        if (nt + 8 < HH / 16)
            __builtin_prefetch(Whb + (size_t)(n + 128) * HH, 0, 0);

        v8f acc0 = {}; v8f acc1 = {};
#pragma clang loop unroll_count(4)
        for (int k0 = 0; k0 < HH; k0 += 32) {
            // ---- B operand: contiguous run of 16 bf16 K-values for this lane
            const int kb = k0 + kcB;
            v8bf b0 = *(const v8bf*)(wr + kb);
            v8bf b1 = *(const v8bf*)(wr + kb + 8);
            v16bf bm;
            for (int e = 0; e < 8; ++e) { bm[e] = b0[e]; bm[8 + e] = b1[e]; }

            // ---- A operands: contiguous run of 16 K-values per M-tile row (matches B)
            const int ka = k0 + kcA;
            v8bf l0 = *(const v8bf*)(a0p + ka);
            v8bf h0 = *(const v8bf*)(a0p + ka + 8);
            v8bf l1 = *(const v8bf*)(a1p + ka);
            v8bf h1 = *(const v8bf*)(a1p + ka + 8);
            v16bf am0, am1;
            for (int e = 0; e < 8; ++e) {
                am0[e] = l0[e]; am0[8 + e] = h0[e];
                am1[e] = l1[e]; am1[8 + e] = h1[e];
            }

            acc0 = __builtin_amdgcn_wmma_f32_16x16x32_bf16(false, am0, false, bm,
                                                           (short)0, acc0, false, false);
            acc1 = __builtin_amdgcn_wmma_f32_16x16x32_bf16(false, am1, false, bm,
                                                           (short)0, acc1, false, false);
        }

        // Fused epilogue: lane's N is fixed -> scalar dec/Wc/V loads broadcast.
        const float dn  = dec_feat[(size_t)b * HH + n];
        const float wcn = W_c[n];
        const float vn  = Vw[n];
        for (int v = 0; v < 8; ++v) {
            part0[v] += vn * tanhf(acc0[v] + dn + cov0[v] * wcn);
            part1[v] += vn * tanhf(acc1[v] + dn + cov1[v] * wcn);
        }
    }

    // Reduce over N within each 16-lane half (lanes share the same 8 M rows).
    for (int off = 8; off >= 1; off >>= 1) {
        for (int v = 0; v < 8; ++v) {
            part0[v] += __shfl_xor(part0[v], off, 16);
            part1[v] += __shfl_xor(part1[v], off, 16);
        }
    }
    if (mrow == 0) {
        for (int v = 0; v < 8; ++v) {
            e_part[wave][mb + v]      = part0[v];
            e_part[wave][16 + mb + v] = part1[v];
        }
    }
    __syncthreads();
    if (tid < TILE_M) {
        float s = 0.f;
        for (int w = 0; w < 8; ++w) s += e_part[w][tid];
        e_out[(size_t)b * SS + s0 + tid] = s;
    }
}

// ---------------------------------------------------------------------------
// Kernel 3: per-batch softmax over S; writes atten_dist and new_coverage,
// and zeroes context (so kernel 4 can accumulate atomically). grid(B), 256.
// ---------------------------------------------------------------------------
__global__ void softmax_kernel(const float* __restrict__ e,
                               const float* __restrict__ mask,
                               const float* __restrict__ coverage,
                               float* __restrict__ dist,
                               float* __restrict__ newcov,
                               float* __restrict__ ctx) {
    __shared__ float sm[SS];
    __shared__ float red[256];
    const int b = blockIdx.x, tid = threadIdx.x;
    const size_t base = (size_t)b * SS;

    float mx = -INFINITY;
    for (int i = tid; i < SS; i += 256) {
        float x = e[base + i] + mask[base + i];
        sm[i] = x;
        mx = fmaxf(mx, x);
    }
    red[tid] = mx; __syncthreads();
    for (int off = 128; off >= 1; off >>= 1) {
        if (tid < off) red[tid] = fmaxf(red[tid], red[tid + off]);
        __syncthreads();
    }
    mx = red[0]; __syncthreads();

    float sum = 0.f;
    for (int i = tid; i < SS; i += 256) {
        float ex = expf(sm[i] - mx);
        sm[i] = ex;
        sum += ex;
    }
    red[tid] = sum; __syncthreads();
    for (int off = 128; off >= 1; off >>= 1) {
        if (tid < off) red[tid] += red[tid + off];
        __syncthreads();
    }
    const float inv = 1.0f / red[0];

    for (int i = tid; i < SS; i += 256) {
        float d = sm[i] * inv;
        dist[base + i]   = d;
        newcov[base + i] = coverage[base + i] + d;
    }
    for (int i = tid; i < HH; i += 256) ctx[(size_t)b * HH + i] = 0.0f;
}

// ---------------------------------------------------------------------------
// Kernel 4: context[b,h] = sum_s dist[b,s] * enc[b,s,h].
// grid(S/256, B), block 256: each block handles a 256-row S-chunk, each
// thread one float4 H-column -> fully coalesced streams, fp32 global atomics.
// ---------------------------------------------------------------------------
__global__ void context_kernel(const float* __restrict__ enc,
                               const float* __restrict__ dist,
                               float* __restrict__ ctx) {
    __shared__ float a[256];
    const int b = blockIdx.y, s0 = blockIdx.x * 256, tid = threadIdx.x;
    a[tid] = dist[(size_t)b * SS + s0 + tid];
    __syncthreads();

    const float4* ep = (const float4*)(enc + ((size_t)b * SS + s0) * HH);
    float4 acc = make_float4(0.f, 0.f, 0.f, 0.f);
    for (int s = 0; s < 256; ++s) {
        float4 v = ep[(size_t)s * (HH / 4) + tid];
        const float w = a[s];
        acc.x += w * v.x; acc.y += w * v.y; acc.z += w * v.z; acc.w += w * v.w;
    }
    float* c = ctx + (size_t)b * HH + tid * 4;
    unsafeAtomicAdd(c + 0, acc.x);
    unsafeAtomicAdd(c + 1, acc.y);
    unsafeAtomicAdd(c + 2, acc.z);
    unsafeAtomicAdd(c + 3, acc.w);
}

// ---------------------------------------------------------------------------
extern "C" void kernel_launch(void* const* d_in, const int* in_sizes, int n_in,
                              void* d_out, int out_size, void* d_ws, size_t ws_size,
                              hipStream_t stream) {
    const float* enc        = (const float*)d_in[0];  // [B,S,H]
    const float* dec_hidden = (const float*)d_in[1];  // [B,H]
    const float* mask       = (const float*)d_in[2];  // [B,S]
    const float* coverage   = (const float*)d_in[3];  // [B,S]
    const float* W_h        = (const float*)d_in[4];  // [H,H]
    const float* W_s        = (const float*)d_in[5];  // [H,H]
    const float* b_s        = (const float*)d_in[6];  // [H]
    const float* W_c        = (const float*)d_in[7];  // [H,1]
    const float* Vw         = (const float*)d_in[8];  // [1,H]

    float* out_ctx  = (float*)d_out;                  // [B,H]
    float* out_dist = out_ctx + BB * HH;              // [B,S]
    float* out_cov  = out_dist + BB * SS;              // [B,S]

    float*  ws_dec = (float*)d_ws;                    // [B,H]
    float*  ws_e   = ws_dec + BB * HH;                // [B,S]
    __bf16* ws_whb = (__bf16*)(ws_e + BB * SS);       // [H,H] bf16

    wh_bf16_kernel<<<dim3((HH * HH) / (256 * 8)), 256, 0, stream>>>(W_h, ws_whb);
    dec_feature_kernel<<<dim3(BB, HH / 256), 256, 0, stream>>>(dec_hidden, W_s, b_s, ws_dec);
    attn_scores_kernel<<<dim3(SS / TILE_M, BB), 256, 0, stream>>>(enc, ws_dec, coverage,
                                                                  ws_whb, W_c, Vw, ws_e);
    softmax_kernel<<<dim3(BB), 256, 0, stream>>>(ws_e, mask, coverage,
                                                 out_dist, out_cov, out_ctx);
    context_kernel<<<dim3(SS / 256, BB), 256, 0, stream>>>(enc, out_dist, out_ctx);
}
